// Qwen2_5_VisionTransformerWrapper_61460982005997
// MI455X (gfx1250) — matrix operand order repaired and checked by reading, not compiled, not run on hardware
//
#include <hip/hip_runtime.h>
#include <hip/hip_bf16.h>
#include <math.h>

// ---------------------------------------------------------------------------
// Qwen2.5-VL vision tower forward for gfx1250 (MI455X), wave32 + WMMA bf16.
// All GEMMs (qkv/proj/gate/up/down/merger) and both attention matmuls use
// v_wmma_f32_16x16x32_bf16 with fp32 accumulation. fp32 operands are converted
// to bf16 while staging tiles into LDS; GEMM uses double-buffered LDS so
// global->LDS staging of tile k+1 overlaps the WMMA block of tile k.
// ---------------------------------------------------------------------------

#define S    2048
#define H    1280
#define NH   16
#define HD   80
#define WIN  64
#define NW   (S / WIN)
#define II   3420
#define OUTD 3584
#define L    8
#define MERGE 4

typedef __attribute__((ext_vector_type(16))) __bf16 v16bf;
typedef __attribute__((ext_vector_type(8)))  float  v8f;

__device__ __forceinline__ unsigned short f2bf(float f) {
  unsigned u = __float_as_uint(f);
  u = (u + 0x7FFFu + ((u >> 16) & 1u)) >> 16;   // round-to-nearest-even
  return (unsigned short)u;
}

__device__ __forceinline__ v8f v8f_splat(float x) {
  v8f r;
#pragma unroll
  for (int i = 0; i < 8; ++i) r[i] = x;
  return r;
}

__device__ __forceinline__ v8f wmma_bf16(v16bf a, v16bf b, v8f c) {
  return __builtin_amdgcn_wmma_f32_16x16x32_bf16(
      /*neg_a=*/false, a, /*neg_b=*/false, b,
      /*c_mod=*/(short)0, c, /*reuse_a=*/false, /*reuse_b=*/false);
}

// A-matrix fragment 16x32 bf16 (ISA 7.12.2): lanes 0-15 hold K=0..7,16..23 of
// row (lane&15); lanes 16-31 hold K=8..15,24..31.
__device__ __forceinline__ v16bf frag_a_bf16(const unsigned short* s, int ldk,
                                             int row0, int k0, int lane) {
  const unsigned short* p = s + (row0 + (lane & 15)) * ldk + k0 + ((lane >> 4) << 3);
  union { v16bf v; unsigned u[8]; } f;
#pragma unroll
  for (int i = 0; i < 4; ++i) f.u[i]     = *(const unsigned*)(p + 2 * i);
#pragma unroll
  for (int i = 0; i < 4; ++i) f.u[4 + i] = *(const unsigned*)(p + 16 + 2 * i);
  return f.v;
}

// B-matrix fragment 32x16 bf16: lanes 0-15 hold K=0..7,16..23 of col (lane&15);
// lanes 16-31 hold K=8..15,24..31. LDS layout is [col][k] so K runs contiguous.
__device__ __forceinline__ v16bf frag_b_bf16(const unsigned short* s, int ldk,
                                             int col0, int k0, int lane) {
  const unsigned short* p = s + (col0 + (lane & 15)) * ldk + k0 + ((lane >> 4) << 3);
  union { v16bf v; unsigned u[8]; } f;
#pragma unroll
  for (int i = 0; i < 4; ++i) f.u[i]     = *(const unsigned*)(p + 2 * i);
#pragma unroll
  for (int i = 0; i < 4; ++i) f.u[4 + i] = *(const unsigned*)(p + 16 + 2 * i);
  return f.v;
}

// Stage 16 contiguous fp32 -> bf16 into LDS via 4x float4 loads. Every K used
// here is a multiple of 4, so a 4-group is either fully inside or fully
// outside [0,K): no per-element masking, one cndmask-style guard per group.
__device__ __forceinline__ void stage16(unsigned short* sdst, const float* g,
                                        int grow, int glimit, int ldg,
                                        int gcol, int K) {
  const float* rowp = g + (size_t)grow * ldg;
#pragma unroll
  for (int j4 = 0; j4 < 4; ++j4) {
    const int gc = gcol + 4 * j4;
    float4 v = make_float4(0.f, 0.f, 0.f, 0.f);
    if (grow < glimit && gc < K) v = *(const float4*)(rowp + gc);
    sdst[4 * j4 + 0] = f2bf(v.x);
    sdst[4 * j4 + 1] = f2bf(v.y);
    sdst[4 * j4 + 2] = f2bf(v.z);
    sdst[4 * j4 + 3] = f2bf(v.w);
  }
}

// ---------------------------------------------------------------------------
// GEMM: C[M,N] = A[M,K] * W[N,K]^T + bias + (res ? res : 0)
// Block tile 128x128, 8 waves of 32(M)x64(N), K-step 32, double-buffered LDS.
// ---------------------------------------------------------------------------
#define BM 128
#define BN 128
#define BK 32

__global__ __launch_bounds__(256)
void gemm_bf16_wmma(const float* __restrict__ A, const float* __restrict__ W,
                    const float* __restrict__ bias, const float* __restrict__ res,
                    float* __restrict__ C, int M, int N, int K) {
  __shared__ unsigned short sA[2][BM * BK];
  __shared__ unsigned short sB[2][BN * BK];

  const int bm = blockIdx.y * BM;
  const int bn = blockIdx.x * BN;
  const int t = threadIdx.x;
  const int wave = t >> 5;
  const int lane = t & 31;
  const int wm = (wave & 3) * 32;   // wave M offset inside block tile
  const int wn = (wave >> 2) * 64;  // wave N offset inside block tile

  v8f acc[2][4];
#pragma unroll
  for (int i = 0; i < 2; ++i)
#pragma unroll
    for (int j = 0; j < 4; ++j) acc[i][j] = v8f_splat(0.f);

  const int srow = t >> 1;           // 0..127
  const int scol = (t & 1) * 16;     // 0 or 16
  unsigned short* myA0 = &sA[0][srow * BK + scol];
  unsigned short* myA1 = &sA[1][srow * BK + scol];
  unsigned short* myB0 = &sB[0][srow * BK + scol];
  unsigned short* myB1 = &sB[1][srow * BK + scol];

  // prologue: stage tile 0 into buffer 0
  stage16(myA0, A, bm + srow, M, K, scol, K);
  stage16(myB0, W, bn + srow, N, K, scol, K);
  __syncthreads();

  int buf = 0;
  for (int kk = 0; kk < K; kk += BK) {
    // ---- stage next tile into the other buffer (overlaps WMMA below) ----
    const int nxt = kk + BK;
    if (nxt < K) {
      stage16(buf ? myA0 : myA1, A, bm + srow, M, K, nxt + scol, K);
      stage16(buf ? myB0 : myB1, W, bn + srow, N, K, nxt + scol, K);
      if (nxt + BK < K) {   // prefetch the K+2 tile (global_prefetch_b8)
        int pr = bm + srow; if (pr >= M) pr = M - 1;
        int pn = bn + srow; if (pn >= N) pn = N - 1;
        __builtin_prefetch(&A[(size_t)pr * K + nxt + BK + scol], 0, 1);
        __builtin_prefetch(&W[(size_t)pn * K + nxt + BK + scol], 0, 1);
      }
    }

    // ---- WMMA over the current buffer ----
    const unsigned short* cA = sA[buf];
    const unsigned short* cB = sB[buf];
#pragma unroll
    for (int am = 0; am < 2; ++am) {
      v16bf a = frag_a_bf16(cA, BK, wm + am * 16, 0, lane);
#pragma unroll
      for (int bi = 0; bi < 4; ++bi) {
        v16bf b = frag_b_bf16(cB, BK, wn + bi * 16, 0, lane);
        acc[am][bi] = wmma_bf16(a, b, acc[am][bi]);
      }
    }
    __syncthreads();   // next iter overwrites `buf`, reads `buf^1`
    buf ^= 1;
  }

  // ---- epilogue: bias + optional residual ----
#pragma unroll
  for (int am = 0; am < 2; ++am) {
#pragma unroll
    for (int bi = 0; bi < 4; ++bi) {
      const int col = bn + wn + bi * 16 + (lane & 15);
      if (col >= N) continue;
      const float bv = bias ? bias[col] : 0.f;
#pragma unroll
      for (int vg = 0; vg < 8; ++vg) {
        const int row = bm + wm + am * 16 + vg + 8 * (lane >> 4);
        if (row >= M) continue;
        float v = acc[am][bi][vg] + bv;
        if (res) v += res[(size_t)row * N + col];
        C[(size_t)row * N + col] = v;
      }
    }
  }
}

// ---------------------------------------------------------------------------
// Flash attention (windowed or full), one block per (head, 64-row q tile).
// 4 waves; wave w owns q rows 16w..16w+15. RoPE fused into Q/K staging.
// qkv layout: [S][3*H]. Masks are all-pass (additive zero) in this model.
// Staging: thread t owns kv row (t&63) and head-dim half (t>>6)*40, so the
// rotate-half select is wave-uniform and there is no div/mod per element.
// ---------------------------------------------------------------------------
__global__ __launch_bounds__(128)
void attn_flash_wmma(const float* __restrict__ qkv, const float* __restrict__ cosp,
                     const float* __restrict__ sinp, float* __restrict__ o,
                     int fullAttn) {
  __shared__ unsigned short sQ[64 * 96];    // q tile, K padded 80->96, pre-scaled
  __shared__ unsigned short sK[64 * 96];    // k tile, padded
  __shared__ unsigned short sVt[80 * 64];   // v tile transposed: [d][kv_row]
  __shared__ unsigned short sP[4 * 16 * 64];// per-wave softmax probs (bf16)

  const int head = blockIdx.x >> 5;   // /NW
  const int qt   = blockIdx.x & 31;
  const int t    = threadIdx.x;
  const int wave = t >> 5;
  const int lane = t & 31;
  const float scale = 0.1118033988749895f;  // 1/sqrt(80)

  const int srow = t & 63;            // staged row (0..63)
  const int dh   = (t >> 6) * 40;     // head-dim half base (0 or 40)
  const float rsign = (dh == 0) ? -1.f : 1.f;
  const int rshift = (dh == 0) ? 40 : -40;

  // ---- stage Q (RoPE + scale), zero-pad d=80..95 ----
  {
    const int s = qt * 64 + srow;
    const float* base = qkv + (size_t)s * (3 * H) + head * HD;
    const float* cs = cosp + (size_t)s * HD;
    const float* sn = sinp + (size_t)s * HD;
#pragma unroll 8
    for (int j = 0; j < 40; ++j) {
      const int d = dh + j;
      const float v = base[d] * cs[d] + rsign * base[d + rshift] * sn[d];
      sQ[srow * 96 + d] = f2bf(v * scale);
    }
    if (dh == 0) {   // d=80..87 by threads 0..63, 88..95 by 64..127
      sQ[srow * 96 + 80 + 0]  = 0; sQ[srow * 96 + 80 + 1]  = 0;
      sQ[srow * 96 + 80 + 2]  = 0; sQ[srow * 96 + 80 + 3]  = 0;
      sQ[srow * 96 + 80 + 4]  = 0; sQ[srow * 96 + 80 + 5]  = 0;
      sQ[srow * 96 + 80 + 6]  = 0; sQ[srow * 96 + 80 + 7]  = 0;
    } else {
      sQ[srow * 96 + 88 + 0]  = 0; sQ[srow * 96 + 88 + 1]  = 0;
      sQ[srow * 96 + 88 + 2]  = 0; sQ[srow * 96 + 88 + 3]  = 0;
      sQ[srow * 96 + 88 + 4]  = 0; sQ[srow * 96 + 88 + 5]  = 0;
      sQ[srow * 96 + 88 + 6]  = 0; sQ[srow * 96 + 88 + 7]  = 0;
    }
  }

  v8f m_run = v8f_splat(-INFINITY);
  v8f l_run = v8f_splat(0.f);
  v8f oacc[5];
#pragma unroll
  for (int n = 0; n < 5; ++n) oacc[n] = v8f_splat(0.f);

  const int kt0 = fullAttn ? 0 : qt;
  const int kt1 = fullAttn ? NW : qt + 1;

  for (int kt = kt0; kt < kt1; ++kt) {
    __syncthreads();  // previous PV reads of sK/sVt complete
    // ---- stage K (RoPE) and V (transposed) ----
    {
      const int s = kt * 64 + srow;
      const float* kb = qkv + (size_t)s * (3 * H) + H + head * HD;
      const float* vb = qkv + (size_t)s * (3 * H) + 2 * H + head * HD;
      const float* cs = cosp + (size_t)s * HD;
      const float* sn = sinp + (size_t)s * HD;
#pragma unroll 8
      for (int j = 0; j < 40; ++j) {
        const int d = dh + j;
        const float v = kb[d] * cs[d] + rsign * kb[d + rshift] * sn[d];
        sK[srow * 96 + d] = f2bf(v);
        sVt[d * 64 + srow] = f2bf(vb[d]);
      }
      const int pbase = srow * 96 + ((dh == 0) ? 80 : 88);
#pragma unroll
      for (int j = 0; j < 8; ++j) sK[pbase + j] = 0;
    }
    __syncthreads();

    // ---- scores: S = (Q*scale) K^T, 16x64 per wave ----
    v8f sc[4];
#pragma unroll
    for (int j = 0; j < 4; ++j) sc[j] = v8f_splat(0.f);
#pragma unroll
    for (int ks = 0; ks < 3; ++ks) {
      v16bf a = frag_a_bf16(sQ, 96, 16 * wave, ks * 32, lane);
#pragma unroll
      for (int j = 0; j < 4; ++j) {
        v16bf b = frag_b_bf16(sK, 96, 16 * j, ks * 32, lane);
        sc[j] = wmma_bf16(a, b, sc[j]);
      }
    }

    // ---- online softmax. C layout: row = vg + 8*(lane>=16), col = lane&15.
    v8f mt = sc[0];
#pragma unroll
    for (int j = 1; j < 4; ++j)
#pragma unroll
      for (int vg = 0; vg < 8; ++vg) mt[vg] = fmaxf(mt[vg], sc[j][vg]);
#pragma unroll
    for (int off = 1; off < 16; off <<= 1)
#pragma unroll
      for (int vg = 0; vg < 8; ++vg) mt[vg] = fmaxf(mt[vg], __shfl_xor(mt[vg], off));

    v8f m_new, alpha;
#pragma unroll
    for (int vg = 0; vg < 8; ++vg) {
      m_new[vg] = fmaxf(m_run[vg], mt[vg]);
      alpha[vg] = expf(m_run[vg] - m_new[vg]);
    }
    v8f lsum = v8f_splat(0.f);
#pragma unroll
    for (int j = 0; j < 4; ++j)
#pragma unroll
      for (int vg = 0; vg < 8; ++vg) {
        const float p = expf(sc[j][vg] - m_new[vg]);
        sc[j][vg] = p;
        lsum[vg] += p;
      }
#pragma unroll
    for (int off = 1; off < 16; off <<= 1)
#pragma unroll
      for (int vg = 0; vg < 8; ++vg) lsum[vg] += __shfl_xor(lsum[vg], off);
#pragma unroll
    for (int vg = 0; vg < 8; ++vg) {
      l_run[vg] = l_run[vg] * alpha[vg] + lsum[vg];
      m_run[vg] = m_new[vg];
    }
#pragma unroll
    for (int n = 0; n < 5; ++n)
#pragma unroll
      for (int vg = 0; vg < 8; ++vg) oacc[n][vg] *= alpha[vg];

    // ---- write P (bf16) to per-wave LDS region ----
#pragma unroll
    for (int j = 0; j < 4; ++j)
#pragma unroll
      for (int vg = 0; vg < 8; ++vg)
        sP[wave * 1024 + (vg + 8 * (lane >> 4)) * 64 + j * 16 + (lane & 15)] =
            f2bf(sc[j][vg]);
    __syncthreads();

    // ---- O += P * V  (A: 16x64 P, B: 64x80 V via sVt[d][row]) ----
#pragma unroll
    for (int ks = 0; ks < 2; ++ks) {
      v16bf a = frag_a_bf16(sP + wave * 1024, 64, 0, ks * 32, lane);
#pragma unroll
      for (int n = 0; n < 5; ++n) {
        v16bf b = frag_b_bf16(sVt, 64, 16 * n, ks * 32, lane);
        oacc[n] = wmma_bf16(a, b, oacc[n]);
      }
    }
  }

  // ---- normalize and store: o[s][head*80 + d] ----
#pragma unroll
  for (int n = 0; n < 5; ++n) {
    const int d = n * 16 + (lane & 15);
#pragma unroll
    for (int vg = 0; vg < 8; ++vg) {
      const int row = 16 * wave + vg + 8 * (lane >> 4);
      const int s = qt * 64 + row;
      o[(size_t)s * H + head * HD + d] = oacc[n][vg] / l_run[vg];
    }
  }
}

// ---------------------------------------------------------------------------
// Elementwise kernels
// ---------------------------------------------------------------------------
__global__ __launch_bounds__(256)
void rmsnorm_kernel(const float* __restrict__ x, const float* __restrict__ w,
                    float* __restrict__ out, int cols) {
  const int row = blockIdx.x;
  const float* xr = x + (size_t)row * cols;
  float ss = 0.f;
  for (int c = threadIdx.x; c < cols; c += 256) { const float v = xr[c]; ss += v * v; }
  __shared__ float red[256];
  red[threadIdx.x] = ss;
  __syncthreads();
  for (int s = 128; s > 0; s >>= 1) {
    if (threadIdx.x < s) red[threadIdx.x] += red[threadIdx.x + s];
    __syncthreads();
  }
  const float rs = rsqrtf(red[0] / (float)cols + 1e-6f);
  float* orow = out + (size_t)row * cols;
  for (int c = threadIdx.x; c < cols; c += 256) orow[c] = xr[c] * rs * w[c];
}

__global__ __launch_bounds__(256)
void silu_mul_kernel(float* __restrict__ g, const float* __restrict__ u, int n) {
  const int i = blockIdx.x * 256 + threadIdx.x;
  if (i < n) {
    const float x = g[i];
    g[i] = (x / (1.f + expf(-x))) * u[i];
  }
}

__global__ __launch_bounds__(256)
void gelu_kernel(float* __restrict__ p, int n) {
  const int i = blockIdx.x * 256 + threadIdx.x;
  if (i < n) {
    const float x = p[i];
    p[i] = 0.5f * x * (1.f + erff(x * 0.70710678118654752f));
  }
}

// ---------------------------------------------------------------------------
// Host orchestration
// ---------------------------------------------------------------------------
static inline dim3 gemm_grid(int M, int N) {
  return dim3((unsigned)((N + BN - 1) / BN), (unsigned)((M + BM - 1) / BM));
}

extern "C" void kernel_launch(void* const* d_in, const int* in_sizes, int n_in,
                              void* d_out, int out_size, void* d_ws, size_t ws_size,
                              hipStream_t stream) {
  (void)in_sizes; (void)n_in; (void)out_size; (void)ws_size;

  const float* hidden  = (const float*)d_in[0];
  // d_in[1], d_in[2]: attention masks (all-ones in this model -> additive 0)
  const float* cosd    = (const float*)d_in[3];
  const float* sind    = (const float*)d_in[4];
  const float* qkv_w   = (const float*)d_in[5];
  const float* qkv_b   = (const float*)d_in[6];
  const float* proj_w  = (const float*)d_in[7];
  const float* proj_b  = (const float*)d_in[8];
  const float* norm1_w = (const float*)d_in[9];
  const float* norm2_w = (const float*)d_in[10];
  const float* gate_w  = (const float*)d_in[11];
  const float* gate_b  = (const float*)d_in[12];
  const float* up_w    = (const float*)d_in[13];
  const float* up_b    = (const float*)d_in[14];
  const float* down_w  = (const float*)d_in[15];
  const float* down_b  = (const float*)d_in[16];
  const float* ln_q_w  = (const float*)d_in[17];
  const float* m0_w    = (const float*)d_in[18];
  const float* m0_b    = (const float*)d_in[19];
  const float* m2_w    = (const float*)d_in[20];
  const float* m2_b    = (const float*)d_in[21];
  float* out = (float*)d_out;

  // workspace carve-out (floats)
  float* x    = (float*)d_ws;                 // [S,H] residual stream
  float* h    = x + (size_t)S * H;            // [S,H] normed activations
  float* qkvb = h + (size_t)S * H;            // [S,3H]; reused for merger mid
  float* ob   = qkvb + (size_t)S * 3 * H;     // [S,H] attention output
  float* g    = ob + (size_t)S * H;           // [S,I]
  float* u    = g + (size_t)S * II;           // [S,I]

  hipMemcpyAsync(x, hidden, (size_t)S * H * sizeof(float),
                 hipMemcpyDeviceToDevice, stream);

  for (int i = 0; i < L; ++i) {
    const int full = (i == 3 || i == 7) ? 1 : 0;

    // attention block
    rmsnorm_kernel<<<S, 256, 0, stream>>>(x, norm1_w + (size_t)i * H, h, H);
    gemm_bf16_wmma<<<gemm_grid(S, 3 * H), 256, 0, stream>>>(
        h, qkv_w + (size_t)i * 3 * H * H, qkv_b + (size_t)i * 3 * H,
        nullptr, qkvb, S, 3 * H, H);
    attn_flash_wmma<<<NH * NW, 128, 0, stream>>>(qkvb, cosd, sind, ob, full);
    gemm_bf16_wmma<<<gemm_grid(S, H), 256, 0, stream>>>(
        ob, proj_w + (size_t)i * H * H, proj_b + (size_t)i * H,
        x /*residual*/, x, S, H, H);

    // MLP block (SwiGLU)
    rmsnorm_kernel<<<S, 256, 0, stream>>>(x, norm2_w + (size_t)i * H, h, H);
    gemm_bf16_wmma<<<gemm_grid(S, II), 256, 0, stream>>>(
        h, gate_w + (size_t)i * II * H, gate_b + (size_t)i * II, nullptr, g, S, II, H);
    gemm_bf16_wmma<<<gemm_grid(S, II), 256, 0, stream>>>(
        h, up_w + (size_t)i * II * H, up_b + (size_t)i * II, nullptr, u, S, II, H);
    silu_mul_kernel<<<((size_t)S * II + 255) / 256, 256, 0, stream>>>(g, u, S * II);
    gemm_bf16_wmma<<<gemm_grid(S, H), 256, 0, stream>>>(
        g, down_w + (size_t)i * H * II, down_b + (size_t)i * H,
        x /*residual*/, x, S, H, II);
  }

  // patch merger: rmsnorm -> view [S/4, H*4] -> Linear/GELU/Linear
  const int MM = S / MERGE;        // 512
  const int MH = H * MERGE;        // 5120
  rmsnorm_kernel<<<S, 256, 0, stream>>>(x, ln_q_w, h, H);  // h viewed as [MM,MH]
  float* mid = qkvb;               // [512,5120] fits in qkv scratch
  gemm_bf16_wmma<<<gemm_grid(MM, MH), 256, 0, stream>>>(
      h, m0_w, m0_b, nullptr, mid, MM, MH, MH);
  gelu_kernel<<<((size_t)MM * MH + 255) / 256, 256, 0, stream>>>(mid, MM * MH);
  gemm_bf16_wmma<<<gemm_grid(MM, OUTD), 256, 0, stream>>>(
      mid, m2_w, m2_b, nullptr, out, MM, OUTD, MH);
}
